// Attention_12524124635716
// MI455X (gfx1250) — matrix operand order repaired and checked by reading, not compiled, not run on hardware
//
#include <hip/hip_runtime.h>

// ---------------------------------------------------------------------------
// Causal self-attention forward for MI455X (gfx1250, wave32, WMMA).
//
// Roofline: x is 64MB fp32 (HBM floor ~3us @23.3TB/s); total math ~11 GFLOP.
// All 4 GEMMs run on v_wmma_f32_16x16x32_bf16 (f32 accumulate). Q/K/V are
// materialized as bf16 in workspace (6MB -> L2 resident, 192MB L2).
//   ws[0      .. 384KB) : W swizzled into WMMA B-fragment layout (bf16)
//   ws[384KB  .. +2MB ) : Q  [b][t][d] bf16, pre-scaled by C^-0.5
//   ws[+2MB   .. +2MB ) : K  [b][s][d] bf16
//   ws[+2MB   .. +2MB ) : V^T[b][d][s] bf16  (so P*V B-fragments are contiguous)
// ---------------------------------------------------------------------------

typedef __bf16 bf16;
typedef __attribute__((ext_vector_type(16))) __bf16   bf16x16;
typedef __attribute__((ext_vector_type(8)))  float    f32x8;
typedef __attribute__((ext_vector_type(4)))  float    f32x4;
typedef __attribute__((ext_vector_type(4)))  unsigned u32x4;

#define EMBD   1024
#define HDIM   64
#define BATCH  8
#define SEQ    2048
#define L2E    1.44269504088896f
#define QSCALE 0.03125f            /* EMBD^-0.5 = 1/32 */

#define WSW_OFF 0u
#define Q_OFF   393216u            /* 3 * 32ksteps * 4nt * 32lanes * 32B      */
#define K_OFF   (Q_OFF + 2097152u) /* 8*2048*64 * 2B                          */
#define V_OFF   (K_OFF + 2097152u)

#if __has_builtin(__builtin_amdgcn_exp2f)
#define EXP2F(x) __builtin_amdgcn_exp2f(x)
#else
#define EXP2F(x) exp2f(x)
#endif

union Frag {                // one 16x32 A-fragment or 32x16 B-fragment (bf16)
  u32x4   q[2];
  bf16x16 v;
};

__device__ __forceinline__ f32x8 wmma_bf16(bf16x16 a, bf16x16 b, f32x8 c) {
  return __builtin_amdgcn_wmma_f32_16x16x32_bf16(
      /*neg_a=*/false, a, /*neg_b=*/false, b,
      /*c_mod=*/(short)0, c, /*reuse_a=*/false, /*reuse_b=*/false);
}

// 16x f32 -> bf16 fragment; native casts let the backend pair them into
// v_cvt_pk_bf16_f32 instead of manual integer RNE sequences.
__device__ __forceinline__ bf16x16 cvt16(f32x4 a0, f32x4 a1, f32x4 b0, f32x4 b1) {
  bf16x16 r;
  r[0]  = (bf16)a0[0]; r[1]  = (bf16)a0[1]; r[2]  = (bf16)a0[2]; r[3]  = (bf16)a0[3];
  r[4]  = (bf16)a1[0]; r[5]  = (bf16)a1[1]; r[6]  = (bf16)a1[2]; r[7]  = (bf16)a1[3];
  r[8]  = (bf16)b0[0]; r[9]  = (bf16)b0[1]; r[10] = (bf16)b0[2]; r[11] = (bf16)b0[3];
  r[12] = (bf16)b1[0]; r[13] = (bf16)b1[1]; r[14] = (bf16)b1[2]; r[15] = (bf16)b1[3];
  return r;
}

// ---------------------------------------------------------------------------
// Kernel 1: swizzle W[C][D] fp32 -> bf16 B-fragment layout.
// Fragment (kstep, nt, lane): VGPR v holds K = kc + g*8 + 2v + 8*(v>=4) + {0,1},
// N = nt*16 + lane%16  (g = lane/16).  One thread builds one 32B fragment.
// ---------------------------------------------------------------------------
__global__ __launch_bounds__(256) void swz_w(const float* __restrict__ Wq,
                                             const float* __restrict__ Wk,
                                             const float* __restrict__ Wv,
                                             u32x4* __restrict__ wsw) {
  int tid = blockIdx.x * 256 + threadIdx.x;
  if (tid >= 3 * 4096) return;
  int matrix = tid >> 12;
  int r      = tid & 4095;
  int ks     = r >> 7;           // K-step (32 of them over C=1024)
  int nt     = (r >> 5) & 3;     // N tile (D=64 -> 4)
  int lane   = r & 31;
  int g      = lane >> 4;
  int n      = nt * 16 + (lane & 15);
  const float* W = (matrix == 0) ? Wq : (matrix == 1) ? Wk : Wv;
  int k0 = ks * 32 + g * 8;
  Frag f;
#pragma unroll
  for (int v = 0; v < 8; ++v) {
    int k = k0 + 2 * v + ((v >> 2) << 3);
    f.v[2 * v]     = (bf16)W[(size_t)k * HDIM + n];
    f.v[2 * v + 1] = (bf16)W[(size_t)(k + 1) * HDIM + n];
  }
  u32x4* dst = wsw + (size_t)(((matrix * 32 + ks) * 4 + nt) * 32 + lane) * 2;
  dst[0] = f.q[0];
  dst[1] = f.q[1];
}

// ---------------------------------------------------------------------------
// Kernel 2: projections.  One wave computes a 64-row x 64-col tile of one of
// Q/K/V.  K-loop over C in steps of 32 -> 16 WMMAs / step.  x is converted
// fp32->bf16 on the fly in A-fragment order (two 32B runs per lane).
// ---------------------------------------------------------------------------
__global__ __launch_bounds__(256, 1) void proj(const float* __restrict__ x,
                                               const u32x4* __restrict__ wsw,
                                               bf16* __restrict__ Qs,
                                               bf16* __restrict__ Kb,
                                               bf16* __restrict__ Vt) {
  int warp   = threadIdx.x >> 5;
  int lane   = threadIdx.x & 31;
  int wg     = blockIdx.x * 8 + warp;   // 0..767
  int matrix = wg >> 8;                 // 0..2  (Q, K, V)
  int row0   = (wg & 255) * 64;         // 256 64-row tiles cover 16384 rows
  int g      = lane >> 4;
  int l15    = lane & 15;

  f32x8 acc[4][4];
  const f32x8 z8 = {0.f, 0.f, 0.f, 0.f, 0.f, 0.f, 0.f, 0.f};
#pragma unroll
  for (int mt = 0; mt < 4; ++mt)
#pragma unroll
    for (int nt = 0; nt < 4; ++nt) acc[mt][nt] = z8;

  for (int kc = 0; kc < EMBD; kc += 32) {
    int ks = kc >> 5;
    Frag bfr[4];
#pragma unroll
    for (int nt = 0; nt < 4; ++nt) {
      const u32x4* p = wsw + (size_t)(((matrix * 32 + ks) * 4 + nt) * 32 + lane) * 2;
      bfr[nt].q[0] = p[0];
      bfr[nt].q[1] = p[1];
    }
#pragma unroll
    for (int mt = 0; mt < 4; ++mt) {
      int row = row0 + mt * 16 + l15;
      const float* xp = x + (size_t)row * EMBD + kc + g * 8;
      f32x4 a0 = *(const f32x4*)(xp);        // K = kc + g*8 + 0..3
      f32x4 a1 = *(const f32x4*)(xp + 4);    // K = kc + g*8 + 4..7
      f32x4 b0 = *(const f32x4*)(xp + 16);   // K = kc + 16 + g*8 + 0..3
      f32x4 b1 = *(const f32x4*)(xp + 20);
      bf16x16 av = cvt16(a0, a1, b0, b1);
#pragma unroll
      for (int nt = 0; nt < 4; ++nt)
        acc[mt][nt] = wmma_bf16(av, bfr[nt].v, acc[mt][nt]);
    }
  }

  // Epilogue: C-layout (VGPR i = row i + 8*(lane/16), lane%16 = col)
  float scale = (matrix == 0) ? QSCALE : 1.0f;
  int hi8 = g * 8;
#pragma unroll
  for (int mt = 0; mt < 4; ++mt)
#pragma unroll
    for (int j = 0; j < 4; ++j)
#pragma unroll
      for (int i = 0; i < 8; ++i) {
        int rowg = row0 + mt * 16 + i + hi8;     // = b*SEQ + t
        int d = j * 16 + l15;
        bf16 bv = (bf16)(acc[mt][j][i] * scale);
        if (matrix == 0)      Qs[(size_t)rowg * HDIM + d] = bv;
        else if (matrix == 1) Kb[(size_t)rowg * HDIM + d] = bv;
        else {
          int bb = rowg >> 11, t = rowg & (SEQ - 1);
          Vt[((size_t)(bb * HDIM + d) << 11) + t] = bv;   // transposed
        }
      }
}

// ---------------------------------------------------------------------------
// Kernel 3: flash attention.  One wave = 32 query rows.  Per 32-key block:
// 8 WMMAs QK^T, online softmax (shfl_xor row reductions, exp2), P staged
// through a private LDS slab to swap C-layout -> A-layout, 8 WMMAs P*V.
// ---------------------------------------------------------------------------
__global__ __launch_bounds__(256, 1) void attn(const bf16* __restrict__ Qs,
                                               const bf16* __restrict__ Kb,
                                               const bf16* __restrict__ Vt,
                                               float* __restrict__ out) {
  __shared__ __align__(16) bf16 pbuf[8][32 * 32];   // 2KB per wave
  int warp = threadIdx.x >> 5;
  int lane = threadIdx.x & 31;
  int wg   = blockIdx.x * 8 + warp;    // 0..511
  int b    = wg >> 6;                  // batch
  int t0   = (wg & 63) * 32;           // query tile base
  int g    = lane >> 4;
  int l15  = lane & 15;
  int hi8  = g * 8;
  bf16* myp = &pbuf[warp][0];

  // Resident Q A-fragments: 2 row tiles x 2 d-chunks (d 0..31, 32..63)
  Frag qf[2][2];
#pragma unroll
  for (int mt = 0; mt < 2; ++mt) {
    const bf16* qp = Qs + (size_t)(b * SEQ + t0 + mt * 16 + l15) * HDIM;
#pragma unroll
    for (int dk = 0; dk < 2; ++dk) {
      qf[mt][dk].q[0] = *(const u32x4*)(qp + dk * 32 + g * 8);
      qf[mt][dk].q[1] = *(const u32x4*)(qp + dk * 32 + 16 + g * 8);
    }
  }

  const f32x8 z8 = {0.f, 0.f, 0.f, 0.f, 0.f, 0.f, 0.f, 0.f};
  f32x8 acc[2][4];
  float mrow[2][8], lrow[2][8];
#pragma unroll
  for (int mt = 0; mt < 2; ++mt) {
#pragma unroll
    for (int j = 0; j < 4; ++j) acc[mt][j] = z8;
#pragma unroll
    for (int i = 0; i < 8; ++i) { mrow[mt][i] = -1e30f; lrow[mt][i] = 0.f; }
  }

  for (int s0 = 0; s0 <= t0; s0 += 32) {
    if (s0 + 32 <= t0) {   // prefetch next K rows / V^T columns into cache
      __builtin_prefetch(Kb + (size_t)(b * SEQ + s0 + 32 + l15) * HDIM + g * 32, 0, 0);
      __builtin_prefetch(Vt + ((size_t)(b * HDIM + lane) << 11) + s0 + 32, 0, 0);
    }
    // K B-fragments: lane -> key s, VGPR -> d (two 16B runs per lane: g*8, 16+g*8)
    Frag kf[2][2];
#pragma unroll
    for (int nt = 0; nt < 2; ++nt) {
      const bf16* kp = Kb + (size_t)(b * SEQ + s0 + nt * 16 + l15) * HDIM;
#pragma unroll
      for (int dk = 0; dk < 2; ++dk) {
        kf[nt][dk].q[0] = *(const u32x4*)(kp + dk * 32 + g * 8);
        kf[nt][dk].q[1] = *(const u32x4*)(kp + dk * 32 + g * 8 + 16);
      }
    }
    // S = Q K^T (already scaled; Q was pre-scaled by C^-0.5)
    f32x8 sc[2][2];
#pragma unroll
    for (int mt = 0; mt < 2; ++mt)
#pragma unroll
      for (int nt = 0; nt < 2; ++nt) {
        f32x8 s = z8;
        s = wmma_bf16(qf[mt][0].v, kf[nt][0].v, s);
        s = wmma_bf16(qf[mt][1].v, kf[nt][1].v, s);
        sc[mt][nt] = s;
      }
    if (s0 == t0) {   // causal mask only on the diagonal block
#pragma unroll
      for (int mt = 0; mt < 2; ++mt)
#pragma unroll
        for (int nt = 0; nt < 2; ++nt)
#pragma unroll
          for (int i = 0; i < 8; ++i) {
            int t = t0 + mt * 16 + i + hi8;
            int s = s0 + nt * 16 + l15;
            sc[mt][nt][i] = (s > t) ? -1e30f : sc[mt][nt][i];
          }
    }
    // Online softmax per row tile
#pragma unroll
    for (int mt = 0; mt < 2; ++mt) {
      float cm[8], fac[8];
#pragma unroll
      for (int i = 0; i < 8; ++i) {
        float v = fmaxf(sc[mt][0][i], sc[mt][1][i]);
        v = fmaxf(v, __shfl_xor(v, 1, 32));
        v = fmaxf(v, __shfl_xor(v, 2, 32));
        v = fmaxf(v, __shfl_xor(v, 4, 32));
        v = fmaxf(v, __shfl_xor(v, 8, 32));   // row max over 16-lane half
        float nm = fmaxf(mrow[mt][i], v);
        fac[i] = EXP2F((mrow[mt][i] - nm) * L2E);
        mrow[mt][i] = nm;
        cm[i] = nm;
      }
#pragma unroll
      for (int nt = 0; nt < 2; ++nt)
#pragma unroll
        for (int i = 0; i < 8; ++i) {
          float p = EXP2F((sc[mt][nt][i] - cm[i]) * L2E);
          sc[mt][nt][i] = p;
          myp[(mt * 16 + i + hi8) * 32 + nt * 16 + l15] = (bf16)p;
        }
#pragma unroll
      for (int i = 0; i < 8; ++i) {
        float s = sc[mt][0][i] + sc[mt][1][i];
        s += __shfl_xor(s, 1, 32);
        s += __shfl_xor(s, 2, 32);
        s += __shfl_xor(s, 4, 32);
        s += __shfl_xor(s, 8, 32);
        lrow[mt][i] = lrow[mt][i] * fac[i] + s;
      }
#pragma unroll
      for (int j = 0; j < 4; ++j)
#pragma unroll
        for (int i = 0; i < 8; ++i) acc[mt][j][i] *= fac[i];
    }
    // V B-fragments from V^T (two 16B runs per lane: g*8, 16+g*8)
    Frag vf[4];
#pragma unroll
    for (int j = 0; j < 4; ++j) {
      const bf16* vp = Vt + ((size_t)(b * HDIM + j * 16 + l15) << 11) + s0 + g * 8;
      vf[j].q[0] = *(const u32x4*)(vp);
      vf[j].q[1] = *(const u32x4*)(vp + 16);
    }
    // P A-fragments from LDS (ds_load_b128 x2), then P*V
#pragma unroll
    for (int mt = 0; mt < 2; ++mt) {
      Frag pf;
      const bf16* pp = myp + (mt * 16 + l15) * 32 + g * 8;
      pf.q[0] = *(const u32x4*)(pp);        // kk = g*8  + 0..7
      pf.q[1] = *(const u32x4*)(pp + 16);   // kk = 16 + g*8 + 0..7
#pragma unroll
      for (int j = 0; j < 4; ++j)
        acc[mt][j] = wmma_bf16(pf.v, vf[j].v, acc[mt][j]);
    }
  }

  // Epilogue: out = acc / l
#pragma unroll
  for (int mt = 0; mt < 2; ++mt) {
    float inv[8];
#pragma unroll
    for (int i = 0; i < 8; ++i) inv[i] = __builtin_amdgcn_rcpf(lrow[mt][i]);
#pragma unroll
    for (int j = 0; j < 4; ++j)
#pragma unroll
      for (int i = 0; i < 8; ++i) {
        int t = t0 + mt * 16 + i + hi8;
        out[((size_t)(b * SEQ + t)) * HDIM + j * 16 + l15] = acc[mt][j][i] * inv[i];
      }
  }
}

// ---------------------------------------------------------------------------
extern "C" void kernel_launch(void* const* d_in, const int* in_sizes, int n_in,
                              void* d_out, int out_size, void* d_ws, size_t ws_size,
                              hipStream_t stream) {
  const float* x  = (const float*)d_in[0];
  const float* Wq = (const float*)d_in[1];
  const float* Wk = (const float*)d_in[2];
  const float* Wv = (const float*)d_in[3];
  float* out = (float*)d_out;
  char* ws = (char*)d_ws;
  u32x4* wsw = (u32x4*)(ws + WSW_OFF);
  bf16* Qs = (bf16*)(ws + Q_OFF);
  bf16* Kb = (bf16*)(ws + K_OFF);
  bf16* Vt = (bf16*)(ws + V_OFF);

  swz_w<<<48, 256, 0, stream>>>(Wq, Wk, Wv, wsw);            // 12288 fragments
  proj<<<96, 256, 0, stream>>>(x, wsw, Qs, Kb, Vt);          // 768 wave-tiles
  attn<<<64, 256, 0, stream>>>(Qs, Kb, Vt, out);             // 512 query tiles
}
